// ResGNN_77025943486564
// MI455X (gfx1250) — matrix operand order repaired and checked by reading, not compiled, not run on hardware
//
#include <hip/hip_runtime.h>
#include <stdint.h>
#include <stddef.h>

// ---------------------------------------------------------------------------
// ResGNN forward for MI455X (gfx1250, wave32).
//   - GEMMs: bf16x3 split-precision WMMA (v_wmma_f32_16x16x32_bf16).
//       * weights pre-converted once to transposed [n][k] bf16 hi/lo ->
//         B fragments are direct 32B global loads (L2-resident, no LDS)
//       * A tile double-buffered in LDS (128-bit ds ops), 1 barrier/K-step,
//         next tile prefetched into VGPRs during compute
//   - GCN propagation: float4 gathers + global_atomic_add_f32 scatters
//     (entire working set < 192MB L2 -> L2-bandwidth bound by design).
// ---------------------------------------------------------------------------

#define NB    2
#define NV    20000
#define NE    320000
#define FIN   32
#define NCOND 32
#define NH    256
#define NENC  128
#define GN_EPS 1e-5f

typedef __attribute__((ext_vector_type(16))) __bf16 v16bf;
typedef __attribute__((ext_vector_type(8)))  __bf16 v8bf;
typedef __attribute__((ext_vector_type(8)))  float  v8f;

#define TB 256        // 8 waves of 32
#define BM 64
#define BN 64
#define LDSS 40       // bf16 row stride: 80B rows -> every 8-elem chunk 16B aligned

__device__ __forceinline__ void split_bf16(float v, __bf16& hi, __bf16& lo) {
  hi = (__bf16)v;
  lo = (__bf16)(v - (float)hi);
}

// C[M,N] = A[M,K1] (++ A2[m%NV, K2]) @ W  (+bias)(+res)(relu)
// Wt: pre-transposed bf16 weights, hi at Wt[n*Ktot+k], lo at Wt[loOff + n*Ktot+k]
// Requires: M % 64 == 0, K1 % 32 == 0, (K1+K2) % 32 == 0  (true for all calls)
__global__ __launch_bounds__(TB) void gemm_bf16x3(
    const float* __restrict__ A, const float* __restrict__ A2,
    const __bf16* __restrict__ Wt, int loOff,
    const float* __restrict__ bias, const float* __restrict__ res,
    float* __restrict__ C,
    int M, int N, int K1, int K2, int doRelu)
{
  __shared__ __bf16 Ah[2 * BM * LDSS];
  __shared__ __bf16 Al[2 * BM * LDSS];

  const int tid  = threadIdx.x;
  const int lane = tid & 31;
  const int wave = tid >> 5;
  const int wm   = wave >> 2;            // 0..1 : 32-row half of 64-row block
  const int wn   = wave & 3;             // 0..3 : 16-col slice of 64-col block
  const int blockM = blockIdx.y * BM;
  const int blockN = blockIdx.x * BN;
  const int Ktot = K1 + K2;
  const int KT   = Ktot >> 5;

  // A staging: 64 rows x 32 k, 8 contiguous fp32 per thread (two b128 loads)
  const int aRow = tid >> 2;             // 0..63
  const int aCol = (tid & 3) * 8;        // 0,8,16,24
  const float* aBase1 = A + (size_t)(blockM + aRow) * K1 + aCol;
  const float* aBase2 = A2 ? (A2 + (size_t)((blockM + aRow) % NV) * K2 + aCol)
                           : nullptr;

  float ar[8];
  auto loadA = [&](int kt) {
    const int k0 = kt << 5;
    const float* p = (k0 < K1) ? (aBase1 + k0) : (aBase2 + (k0 - K1));
    #pragma unroll
    for (int j = 0; j < 8; ++j) ar[j] = p[j];
  };

  // B fragment base: each lane owns one n-column, contiguous k (32B loads)
  const int bk = (lane >> 4) * 16;
  const size_t wrow = (size_t)(blockN + wn * 16 + (lane & 15)) * Ktot;
  const __bf16* wh = Wt + wrow + bk;
  const __bf16* wl = Wt + loOff + wrow + bk;

  const int am = lane & 15;
  const int ak = (lane >> 4) * 16;

  v8f acc0 = {};
  v8f acc1 = {};

  loadA(0);
  int buf = 0;
  for (int kt = 0; kt < KT; ++kt) {
    // convert + store prefetched A tile (packed 128-bit LDS stores)
    {
      v8bf hv, lv;
      #pragma unroll
      for (int j = 0; j < 8; ++j) {
        __bf16 h, l; split_bf16(ar[j], h, l);
        hv[j] = h; lv[j] = l;
      }
      const int off = buf * (BM * LDSS) + aRow * LDSS + aCol;
      *(v8bf*)(Ah + off) = hv;
      *(v8bf*)(Al + off) = lv;
    }
    __syncthreads();
    if (kt + 1 < KT) loadA(kt + 1);      // overlap next global fetch with math

    const int k0 = kt << 5;
    const v16bf bh = *(const v16bf*)(wh + k0);
    const v16bf bl = *(const v16bf*)(wl + k0);

    const int rb = buf * (BM * LDSS);
    #pragma unroll
    for (int sub = 0; sub < 2; ++sub) {
      const int row = rb + (wm * 32 + sub * 16 + am) * LDSS;
      const v8bf a0h = *(const v8bf*)(Ah + row + ak);
      const v8bf a1h = *(const v8bf*)(Ah + row + 8 + ak);
      const v8bf a0l = *(const v8bf*)(Al + row + ak);
      const v8bf a1l = *(const v8bf*)(Al + row + 8 + ak);
      const v16bf ah = __builtin_shufflevector(a0h, a1h,
          0,1,2,3,4,5,6,7,8,9,10,11,12,13,14,15);
      const v16bf al = __builtin_shufflevector(a0l, a1l,
          0,1,2,3,4,5,6,7,8,9,10,11,12,13,14,15);
      v8f acc = sub ? acc1 : acc0;
      // bf16x3: hi*hi + hi*lo + lo*hi ~= fp32 product
      acc = __builtin_amdgcn_wmma_f32_16x16x32_bf16(false, ah, false, bh, (short)0, acc, false, false);
      acc = __builtin_amdgcn_wmma_f32_16x16x32_bf16(false, ah, false, bl, (short)0, acc, false, false);
      acc = __builtin_amdgcn_wmma_f32_16x16x32_bf16(false, al, false, bh, (short)0, acc, false, false);
      if (sub) acc1 = acc; else acc0 = acc;
    }
    buf ^= 1;
  }

  // epilogue: D layout — VGPR r: lanes 0-15 M=r, lanes 16-31 M=r+8
  const int half = lane >> 4;
  const int nloc = lane & 15;
  #pragma unroll
  for (int sub = 0; sub < 2; ++sub) {
    v8f acc = sub ? acc1 : acc0;
    #pragma unroll
    for (int r = 0; r < 8; ++r) {
      const int m = blockM + wm * 32 + sub * 16 + half * 8 + r;
      const int n = blockN + wn * 16 + nloc;
      if (n < N) {
        float v = acc[r];
        if (bias)   v += bias[n];
        if (res)    v += res[(size_t)m * N + n];
        if (doRelu) v = fmaxf(v, 0.f);
        C[(size_t)m * N + n] = v;
      }
    }
  }
}

// fp32 W[K,N] -> transposed bf16 hi/lo Wt[n*K+k], Wt[loOff + n*K+k]
__global__ void wconv_k(const float* __restrict__ W, __bf16* __restrict__ out,
                        int K, int N, int loOff) {
  const int idx = blockIdx.x * blockDim.x + threadIdx.x;
  if (idx >= N * K) return;
  const int n = idx / K;
  const int k = idx % K;
  __bf16 hi, lo;
  split_bf16(W[(size_t)k * N + n], hi, lo);
  out[(size_t)n * K + k] = hi;
  out[(size_t)loOff + (size_t)n * K + k] = lo;
}

// ---------------------------------------------------------------------------
// prep + propagation + pointwise kernels
// ---------------------------------------------------------------------------

__global__ void fill_k(float* __restrict__ p, float v, size_t n) {
  size_t i = (size_t)blockIdx.x * blockDim.x + threadIdx.x;
  if (i < n) p[i] = v;
}

__global__ void deg_accum_k(float* __restrict__ deg, const int* __restrict__ dst) {
  int e = blockIdx.x * blockDim.x + threadIdx.x;
  if (e < NE) atomicAdd(&deg[dst[e]], 1.f);
}

__global__ void rsqrt_k(float* __restrict__ d) {
  int v = blockIdx.x * blockDim.x + threadIdx.x;
  if (v < NV) d[v] = rsqrtf(fmaxf(d[v], 1.f));
}

__global__ void edge_norm_k(float* __restrict__ enorm, const float* __restrict__ dinv,
                            const int* __restrict__ src, const int* __restrict__ dst) {
  int e = blockIdx.x * blockDim.x + threadIdx.x;
  if (e < NE) enorm[e] = dinv[src[e]] * dinv[dst[e]];
}

// out[b,v,:] = dinv[v]^2 * h[b,v,:] + bias   (self-loop term + bias, full overwrite)
__global__ __launch_bounds__(256) void selfloop_bias_k(
    float* __restrict__ out, const float* __restrict__ h,
    const float* __restrict__ dinv, const float* __restrict__ bias,
    int F, int qsh, size_t total)
{
  const size_t idx = (size_t)blockIdx.x * 256 + threadIdx.x;
  if (idx >= total) return;
  const size_t row = idx >> qsh;                 // b*NV + v
  const int q = (int)(idx & ((1u << qsh) - 1));
  const int v = (int)(row % (size_t)NV);
  const float di = dinv[v];
  const float sw = di * di;
  const float4 hv = *(const float4*)(h + row * F + q * 4);
  float4 o;
  o.x = sw * hv.x + bias[q * 4 + 0];
  o.y = sw * hv.y + bias[q * 4 + 1];
  o.z = sw * hv.z + bias[q * 4 + 2];
  o.w = sw * hv.w + bias[q * 4 + 3];
  *(float4*)(out + row * F + q * 4) = o;
}

// out[b,dst,:] += enorm[e] * h[b,src,:]   (float4 gather, 4x f32 atomics)
__global__ __launch_bounds__(256) void edge_scatter_k(
    float* __restrict__ out, const float* __restrict__ h,
    const int* __restrict__ src, const int* __restrict__ dst,
    const float* __restrict__ enorm, int F, int qsh)
{
  const unsigned idx = blockIdx.x * 256u + threadIdx.x;
  const unsigned total = (unsigned)NE << qsh;
  if (idx >= total) return;
  const unsigned e = idx >> qsh;
  const unsigned q = idx & ((1u << qsh) - 1);
  const float wgt = enorm[e];
  const int s = src[e];
  const int d = dst[e];
  #pragma unroll
  for (int b = 0; b < NB; ++b) {
    const float4 hv = *(const float4*)(h + ((size_t)(b * NV + s) * F) + q * 4u);
    float* o = out + ((size_t)(b * NV + d) * F) + q * 4u;
    atomicAdd(o + 0, wgt * hv.x);
    atomicAdd(o + 1, wgt * hv.y);
    atomicAdd(o + 2, wgt * hv.z);
    atomicAdd(o + 3, wgt * hv.w);
  }
}

__global__ void relu_k(float* __restrict__ p, size_t n) {
  size_t i = (size_t)blockIdx.x * blockDim.x + threadIdx.x;
  if (i < n) p[i] = fmaxf(p[i], 0.f);
}

__global__ void add_in_k(float* __restrict__ out, const float* __restrict__ a, size_t n) {
  size_t i = (size_t)blockIdx.x * blockDim.x + threadIdx.x;
  if (i < n) out[i] += a[i];
}

// GroupNorm(1,C): per-batch sum / sumsq reduction (blocks never straddle a batch)
__global__ __launch_bounds__(256) void gn_partial_k(const float* __restrict__ x,
                                                    float* __restrict__ stat) {
  __shared__ float s1[256];
  __shared__ float s2[256];
  const size_t idx = (size_t)blockIdx.x * 256 + threadIdx.x;
  const int b = (int)(idx / (size_t)(NV * NH));
  const float v = x[idx];
  s1[threadIdx.x] = v;
  s2[threadIdx.x] = v * v;
  __syncthreads();
  #pragma unroll
  for (int s = 128; s > 0; s >>= 1) {
    if ((int)threadIdx.x < s) {
      s1[threadIdx.x] += s1[threadIdx.x + s];
      s2[threadIdx.x] += s2[threadIdx.x + s];
    }
    __syncthreads();
  }
  if (threadIdx.x == 0) {
    atomicAdd(&stat[b * 2 + 0], s1[0]);
    atomicAdd(&stat[b * 2 + 1], s2[0]);
  }
}

__global__ __launch_bounds__(256) void gn_apply_k(
    const float* __restrict__ x, float* __restrict__ y,
    const float* __restrict__ stat, const float* __restrict__ gamma,
    const float* __restrict__ beta)
{
  const size_t idx = (size_t)blockIdx.x * 256 + threadIdx.x;
  const int b = (int)(idx / (size_t)(NV * NH));
  const int c = (int)(idx & (NH - 1));
  const float inv = 1.f / (float)(NV * NH);
  const float mu  = stat[b * 2 + 0] * inv;
  const float var = stat[b * 2 + 1] * inv - mu * mu;
  y[idx] = (x[idx] - mu) * rsqrtf(var + GN_EPS) * gamma[c] + beta[c];
}

// gb[b, 0:2H] = t_embed[b] @ film_w + film_b   (tiny: 1024 dots of length 128)
__global__ void film_gemm_k(const float* __restrict__ t, const float* __restrict__ fw,
                            const float* __restrict__ fb, float* __restrict__ gb) {
  const int i = blockIdx.x * blockDim.x + threadIdx.x;
  if (i >= NB * 2 * NH) return;
  const int b = i / (2 * NH);
  const int j = i % (2 * NH);
  float s = fb[j];
  #pragma unroll 4
  for (int k = 0; k < NENC; ++k) s += t[b * NENC + k] * fw[k * (2 * NH) + j];
  gb[i] = s;
}

// h = relu(h * gamma[b,c] + beta[b,c])
__global__ __launch_bounds__(256) void film_apply_k(float* __restrict__ h,
                                                    const float* __restrict__ gb) {
  const size_t idx = (size_t)blockIdx.x * 256 + threadIdx.x;
  const int b = (int)(idx / (size_t)(NV * NH));
  const int c = (int)(idx & (NH - 1));
  const float g  = gb[b * 2 * NH + c];
  const float be = gb[b * 2 * NH + NH + c];
  h[idx] = fmaxf(h[idx] * g + be, 0.f);
}

// ---------------------------------------------------------------------------
// orchestration
// ---------------------------------------------------------------------------
extern "C" void kernel_launch(void* const* d_in, const int* in_sizes, int n_in,
                              void* d_out, int out_size, void* d_ws, size_t ws_size,
                              hipStream_t stream) {
  (void)in_sizes; (void)n_in; (void)out_size; (void)ws_size;

  const float* x      = (const float*)d_in[0];   // [B,V,32]
  const float* cond   = (const float*)d_in[1];   // [V,32]
  const int*   eidx   = (const int*)  d_in[2];   // [2,E]
  const float* temb   = (const float*)d_in[3];   // [B,128]
  const float* enc_w  = (const float*)d_in[4];   // [32,256]
  const float* enc_b  = (const float*)d_in[5];
  const float* dec_w  = (const float*)d_in[6];   // [256,32]
  const float* dec_b  = (const float*)d_in[7];
  const int*   src    = eidx;
  const int*   dst    = eidx + NE;

  // workspace carve (256B aligned)
  char* wsp = (char*)d_ws;
  auto carveB = [&](size_t bytes) -> void* {
    void* p = (void*)wsp;
    wsp += (bytes + 255) & ~(size_t)255;
    return p;
  };
  const size_t MT = (size_t)NB * NV;             // 40000 rows (multiple of 64)
  float* dinv  = (float*)carveB((size_t)NV * 4);
  float* enorm = (float*)carveB((size_t)NE * 4);
  float* stat  = (float*)carveB(64);
  float* gb    = (float*)carveB((size_t)NB * 2 * NH * 4);
  float* bufX  = (float*)carveB(MT * NH * 4);    // current x / skip
  float* t0    = (float*)carveB(MT * NH * 4);
  float* t1    = (float*)carveB(MT * NH * 4);

  const size_t nBVH = MT * NH;                   // 10,240,000 (exact /256)
  const int bigGrid = (int)(nBVH / 256);

  // pre-convert each weight matrix once: [K,N] fp32 -> [pad64(N)][K] bf16 hi/lo
  auto prep_w = [&](const float* W, int K, int N) -> const __bf16* {
    const int pad = (N + 63) & ~63;              // rows n>=N read junk, never stored
    __bf16* p = (__bf16*)carveB((size_t)2 * pad * K * sizeof(__bf16));
    const int total = N * K;
    wconv_k<<<(total + 255) / 256, 256, 0, stream>>>(W, p, K, N, pad * K);
    return p;
  };

  auto gemm = [&](const float* A, const float* A2, const __bf16* Wt,
                  const float* bias, const float* res, float* C,
                  int M, int N, int K1, int K2, int relu) {
    const int Ktot = K1 + K2;
    const int loOff = ((N + 63) & ~63) * Ktot;
    dim3 grid((N + BN - 1) / BN, (M + BM - 1) / BM);
    gemm_bf16x3<<<grid, TB, 0, stream>>>(A, A2, Wt, loOff, bias, res, C,
                                         M, N, K1, K2, relu);
  };
  auto gcn_prop = [&](const float* hin, float* out, const float* bias, int F, int qsh) {
    size_t nq = MT * (size_t)(F >> 2);
    selfloop_bias_k<<<(int)((nq + 255) / 256), 256, 0, stream>>>(out, hin, dinv, bias, F, qsh, nq);
    unsigned tot = (unsigned)NE << qsh;
    edge_scatter_k<<<(int)((tot + 255u) / 256u), 256, 0, stream>>>(out, hin, src, dst, enorm, F, qsh);
  };

  // ---- weight conversions (once per launch, all on stream)
  const __bf16* Wenc = prep_w(enc_w, FIN, NH);
  const __bf16* Wdec = prep_w(dec_w, NH, FIN);
  const __bf16 *Wc[2][3], *Wl[2][3];
  const float  *Cb[2][3], *Lb[2][3], *Gma[2], *Bta[2], *Fw[2], *Fb[2];
  for (int bi = 0; bi < 2; ++bi) {
    const int base = 8 + bi * 16;
    Wc[bi][0] = prep_w((const float*)d_in[base + 0], NH + NCOND, NH);
    Wc[bi][1] = prep_w((const float*)d_in[base + 1], NH, NH);
    Wc[bi][2] = prep_w((const float*)d_in[base + 2], NH, NH);
    Cb[bi][0] = (const float*)d_in[base + 3];
    Cb[bi][1] = (const float*)d_in[base + 4];
    Cb[bi][2] = (const float*)d_in[base + 5];
    Wl[bi][0] = prep_w((const float*)d_in[base + 6], NH, NH);
    Wl[bi][1] = prep_w((const float*)d_in[base + 7], NH, NH);
    Wl[bi][2] = prep_w((const float*)d_in[base + 8], NH, NH);
    Lb[bi][0] = (const float*)d_in[base + 9];
    Lb[bi][1] = (const float*)d_in[base + 10];
    Lb[bi][2] = (const float*)d_in[base + 11];
    Gma[bi]   = (const float*)d_in[base + 12];
    Bta[bi]   = (const float*)d_in[base + 13];
    Fw[bi]    = (const float*)d_in[base + 14];
    Fb[bi]    = (const float*)d_in[base + 15];
  }

  // ---- degree / normalization prep (self-loops: deg starts at 1.0)
  fill_k<<<(NV + 255) / 256, 256, 0, stream>>>(dinv, 1.0f, NV);
  deg_accum_k<<<(NE + 255) / 256, 256, 0, stream>>>(dinv, dst);
  rsqrt_k<<<(NV + 255) / 256, 256, 0, stream>>>(dinv);
  edge_norm_k<<<(NE + 255) / 256, 256, 0, stream>>>(enorm, dinv, src, dst);

  // ---- encoder GCN: bufX = propagate(x @ enc_w) + enc_b
  gemm(x, nullptr, Wenc, nullptr, nullptr, t0, (int)MT, NH, FIN, 0, 0);
  gcn_prop(t0, bufX, enc_b, NH, 6);

  // ---- 2 ResGNN blocks
  for (int bi = 0; bi < 2; ++bi) {
    // layer 0: h = relu(gcn(concat(x, cond))); h = relu(h@lw0 + lb0)
    gemm(bufX, cond, Wc[bi][0], nullptr, nullptr, t0, (int)MT, NH, NH, NCOND, 0);
    gcn_prop(t0, t1, Cb[bi][0], NH, 6);
    relu_k<<<bigGrid, 256, 0, stream>>>(t1, nBVH);
    gemm(t1, nullptr, Wl[bi][0], Lb[bi][0], nullptr, t0, (int)MT, NH, NH, 0, 1);

    // layer 1: GroupNorm -> relu(gcn) -> relu(lin)
    fill_k<<<1, 64, 0, stream>>>(stat, 0.f, 4);
    gn_partial_k<<<bigGrid, 256, 0, stream>>>(t0, stat);
    gn_apply_k<<<bigGrid, 256, 0, stream>>>(t0, t1, stat, Gma[bi], Bta[bi]);
    gemm(t1, nullptr, Wc[bi][1], nullptr, nullptr, t0, (int)MT, NH, NH, 0, 0);
    gcn_prop(t0, t1, Cb[bi][1], NH, 6);
    relu_k<<<bigGrid, 256, 0, stream>>>(t1, nBVH);
    gemm(t1, nullptr, Wl[bi][1], Lb[bi][1], nullptr, t0, (int)MT, NH, NH, 0, 1);

    // last conv (no relu), FiLM, relu, lin2 + residual (in place into bufX)
    gemm(t0, nullptr, Wc[bi][2], nullptr, nullptr, t1, (int)MT, NH, NH, 0, 0);
    gcn_prop(t1, t0, Cb[bi][2], NH, 6);
    film_gemm_k<<<(NB * 2 * NH + 255) / 256, 256, 0, stream>>>(temb, Fw[bi], Fb[bi], gb);
    film_apply_k<<<bigGrid, 256, 0, stream>>>(t0, gb);
    gemm(t0, nullptr, Wl[bi][2], Lb[bi][2], bufX, bufX, (int)MT, NH, NH, 0, 0);
  }

  // ---- decoder GCN + input skip
  float* out = (float*)d_out;
  gemm(bufX, nullptr, Wdec, nullptr, nullptr, t0, (int)MT, FIN, NH, 0, 0);
  gcn_prop(t0, out, dec_b, FIN, 3);
  add_in_k<<<(int)((MT * FIN + 255) / 256), 256, 0, stream>>>(out, x, MT * FIN);
}
